// MaskedAutoEncoder_32100585570529
// MI455X (gfx1250) — matrix operand order repaired and checked by reading, not compiled, not run on hardware
//
#include <hip/hip_runtime.h>
#include <hip/hip_bf16.h>
#include <math.h>

// ---------------------------------------------------------------------------
// Model constants (from the reference)
// ---------------------------------------------------------------------------
#define BATCH   8
#define EMB     1024
#define DECD    512
#define NHEAD   16
#define ENC_L   24
#define DEC_L   8
#define TTOK    196
#define KEEPN   49
#define PDIM    768      // C*P*P = 3*16*16
#define ENC_N   50       // cls + 49 kept
#define DEC_N   197      // cls + 196
#define NOGUARD 0x7FFFFFFF

typedef __attribute__((ext_vector_type(16))) __bf16 v16bf;
typedef __attribute__((ext_vector_type(8)))  float  v8f;

// f32 -> bf16 round-to-nearest; two floats -> one dword (lo=a, hi=b).
// (u + 0x8000) >> 16 rounding, then a single v_perm_b32 picks the two high
// halves: result bytes = {ua.b2, ua.b3, ub.b2, ub.b3}.
static __device__ __forceinline__ unsigned pack2bf(float a, float b) {
  union { float f; unsigned u; } x, y; x.f = a; y.f = b;
  unsigned ua = x.u + 0x8000u;
  unsigned ub = y.u + 0x8000u;
  return __builtin_amdgcn_perm(ub, ua, 0x07060302u);
}

// ---------------------------------------------------------------------------
// Stage helpers.  Assumptions guaranteed by every call site:
//   K % 32 == 0,  N % 64 == 0  (only M needs a row guard)
// LDS tiles: T[row][40] bf16, row stride 80 B (16-B aligned at cols 0,8,16,24)
// ---------------------------------------------------------------------------
__device__ __forceinline__ void stage_rowmajor(__bf16 (*Ts)[40], const float* __restrict__ S,
                                               int brow, int k0, int nrows, int ld, int tid) {
  // thread -> (row = tid>>1, 16-col chunk = (tid&1)*16); 64x32 tile
  const int r  = tid >> 1;
  const int c  = (tid & 1) * 16;
  const int gr = brow + r;
  uint4 w0, w1;
  if (gr < nrows) {
    const float4* src = (const float4*)(S + (size_t)gr * ld + k0 + c);
    float4 f0 = src[0], f1 = src[1], f2 = src[2], f3 = src[3];
    w0.x = pack2bf(f0.x, f0.y); w0.y = pack2bf(f0.z, f0.w);
    w0.z = pack2bf(f1.x, f1.y); w0.w = pack2bf(f1.z, f1.w);
    w1.x = pack2bf(f2.x, f2.y); w1.y = pack2bf(f2.z, f2.w);
    w1.z = pack2bf(f3.x, f3.y); w1.w = pack2bf(f3.z, f3.w);
  } else {
    w0.x = w0.y = w0.z = w0.w = 0u;
    w1.x = w1.y = w1.z = w1.w = 0u;
  }
  *(uint4*)&Ts[r][c]     = w0;
  *(uint4*)&Ts[r][c + 8] = w1;
}

// B row-major (K,N): stage transposed into Bs[n][k].  4k x 4n micro-tile per
// thread: 4 vector loads along n, in-register 4x4 transpose, 4x ds_store_b64.
__device__ __forceinline__ void stage_b_notrans(__bf16 (*Bs)[40], const float* __restrict__ Bm,
                                                int bn, int k0, int N, int tid) {
  const int kk = (tid & 7) * 4;        // 0..28
  const int n0 = (tid >> 3) * 4;       // 0..60
  const float* base = Bm + (size_t)(k0 + kk) * N + bn + n0;
  float4 r0 = *(const float4*)(base);
  float4 r1 = *(const float4*)(base + (size_t)N);
  float4 r2 = *(const float4*)(base + 2 * (size_t)N);
  float4 r3 = *(const float4*)(base + 3 * (size_t)N);
  uint2 c0, c1, c2, c3;
  c0.x = pack2bf(r0.x, r1.x); c0.y = pack2bf(r2.x, r3.x);
  c1.x = pack2bf(r0.y, r1.y); c1.y = pack2bf(r2.y, r3.y);
  c2.x = pack2bf(r0.z, r1.z); c2.y = pack2bf(r2.z, r3.z);
  c3.x = pack2bf(r0.w, r1.w); c3.y = pack2bf(r2.w, r3.w);
  *(uint2*)&Bs[n0 + 0][kk] = c0;
  *(uint2*)&Bs[n0 + 1][kk] = c1;
  *(uint2*)&Bs[n0 + 2][kk] = c2;
  *(uint2*)&Bs[n0 + 3][kk] = c3;
}

// ---------------------------------------------------------------------------
// bf16 WMMA GEMM:  C[M,N] (+)= A[M,K] * B(K,N) + bias[N]
//   transB=0 : B row-major (K,N);  transB=1 : B is (N,K) -> use B^T
// 128 threads (4 waves), block tile 64x64, wave tile 32x32 = 2x2 WMMA,
// BK=32, LDS double-buffered (one barrier per K-step).
// ---------------------------------------------------------------------------
__global__ __launch_bounds__(128)
void gemm_bf16_wmma(const float* __restrict__ A, const float* __restrict__ Bm,
                    const float* __restrict__ bias, float* __restrict__ C,
                    int M, int N, int K, int transB, int addC) {
  __shared__ __bf16 Asm[2][64][40];
  __shared__ __bf16 Bsm[2][64][40];

  const int tid  = threadIdx.x;
  const int lane = tid & 31;
  const int wave = tid >> 5;
  const int wm   = (wave >> 1) * 32;
  const int wn   = (wave & 1) * 32;
  const int r16  = lane & 15;
  const int hi   = lane >> 4;
  const int bm   = blockIdx.y * 64;
  const int bn   = blockIdx.x * 64;

  v8f acc[2][2] = {};

  // prologue: stage first K-tile into buffer 0
  stage_rowmajor(Asm[0], A, bm, 0, M, K, tid);
  if (transB) stage_rowmajor(Bsm[0], Bm, bn, 0, NOGUARD, K, tid);
  else        stage_b_notrans(Bsm[0], Bm, bn, 0, N, tid);

  int cur = 0;
  for (int k0 = 0; k0 < K; k0 += 32) {
    __syncthreads();
    const int k1 = k0 + 32;
    if (k1 < K) {
      // stage next tile into the other buffer while computing this one
      stage_rowmajor(Asm[cur ^ 1], A, bm, k1, M, K, tid);
      if (transB) stage_rowmajor(Bsm[cur ^ 1], Bm, bn, k1, NOGUARD, K, tid);
      else        stage_b_notrans(Bsm[cur ^ 1], Bm, bn, k1, N, tid);
      if (k1 + 32 < K) {
        const float* p = transB ? &Bm[(size_t)(bn + (tid >> 1)) * K + (k1 + 32)]
                                : &Bm[(size_t)(k1 + 32 + (tid & 31)) * N + bn];
        __builtin_prefetch(p, 0, 1);   // global_prefetch_b8 on next weight panel
      }
    }

    // fragments in contiguous k order (k = hi*16 + i, same for A and B) (reads merge into ds_load_b128)
    v16bf afr[2], bfr[2];
#pragma unroll
    for (int t = 0; t < 2; ++t) {
      const int ar = wm + t * 16 + r16;
#pragma unroll
      for (int i = 0; i < 16; ++i)
        afr[t][i] = Asm[cur][ar][hi * 16 + i];
      const int bc = wn + t * 16 + r16;
#pragma unroll
      for (int i = 0; i < 16; ++i)
        bfr[t][i] = Bsm[cur][bc][hi * 16 + i];
    }

#pragma unroll
    for (int tm = 0; tm < 2; ++tm)
#pragma unroll
      for (int tn = 0; tn < 2; ++tn)
        acc[tm][tn] = __builtin_amdgcn_wmma_f32_16x16x32_bf16(
            false, afr[tm], false, bfr[tn], (short)0, acc[tm][tn], false, false);

    cur ^= 1;
  }

  // epilogue: C/D layout M = hi*8 + r, N = lane&15 (N bound never needed)
#pragma unroll
  for (int tm = 0; tm < 2; ++tm)
#pragma unroll
    for (int tn = 0; tn < 2; ++tn)
#pragma unroll
      for (int r = 0; r < 8; ++r) {
        int gr = bm + wm + tm * 16 + hi * 8 + r;
        int gc = bn + wn + tn * 16 + r16;
        if (gr < M) {
          float v = acc[tm][tn][r] + bias[gc];
          size_t o = (size_t)gr * N + gc;
          if (addC) C[o] += v; else C[o] = v;
        }
      }
}

// ---------------------------------------------------------------------------
// Elementwise / reduction helpers
// ---------------------------------------------------------------------------
__global__ void patchify_k(const float* __restrict__ img, float* __restrict__ patches) {
  int idx = blockIdx.x * 256 + threadIdx.x;
  const int total = BATCH * TTOK * PDIM;
  if (idx >= total) return;
  int c3 = idx % PDIM;
  int t  = (idx / PDIM) % TTOK;
  int b  = idx / (PDIM * TTOK);
  int c  = c3 / 256;
  int py = (c3 % 256) / 16;
  int px = c3 % 16;
  int gy = t / 14, gx = t % 14;
  patches[idx] = img[(((size_t)b * 3 + c) * 224 + gy * 16 + py) * 224 + gx * 16 + px];
}

__global__ void idsrestore_k(const int* __restrict__ shuf, int* __restrict__ restore) {
  int idx = blockIdx.x * 256 + threadIdx.x;
  if (idx >= BATCH * TTOK) return;
  int b = idx / TTOK, j = idx % TTOK;
  restore[b * TTOK + shuf[idx]] = j;
}

__global__ void enc_assemble_k(const float* __restrict__ embed, const float* __restrict__ cls,
                               const float* __restrict__ pos, const int* __restrict__ shuf,
                               float* __restrict__ x) {
  int idx = blockIdx.x * 256 + threadIdx.x;
  if (idx >= BATCH * ENC_N * EMB) return;
  int e = idx % EMB;
  int n = (idx / EMB) % ENC_N;
  int b = idx / (EMB * ENC_N);
  float v;
  if (n == 0) v = cls[e] + pos[e];
  else {
    int id = shuf[b * TTOK + (n - 1)];
    v = embed[((size_t)b * TTOK + id) * EMB + e] + pos[(size_t)(id + 1) * EMB + e];
  }
  x[idx] = v;
}

__global__ void dec_assemble_k(const float* __restrict__ ydec, const float* __restrict__ mtok,
                               const int* __restrict__ restore, float* __restrict__ x) {
  int idx = blockIdx.x * 256 + threadIdx.x;
  if (idx >= BATCH * DEC_N * DECD) return;
  int e = idx % DECD;
  int n = (idx / DECD) % DEC_N;
  int b = idx / (DECD * DEC_N);
  float v;
  if (n == 0) v = ydec[((size_t)b * ENC_N) * DECD + e];
  else {
    int r = restore[b * TTOK + (n - 1)];
    v = (r < KEEPN) ? ydec[((size_t)b * ENC_N + 1 + r) * DECD + e] : mtok[e];
  }
  x[idx] = v;
}

__global__ void layernorm_k(const float* __restrict__ x, const float* __restrict__ w,
                            const float* __restrict__ b, float* __restrict__ out, int E) {
  __shared__ float sred[256];
  int row = blockIdx.x, tid = threadIdx.x;
  const float* xr = x + (size_t)row * E;
  float s = 0.f;
  for (int i = tid; i < E; i += 256) s += xr[i];
  sred[tid] = s; __syncthreads();
  for (int st = 128; st > 0; st >>= 1) { if (tid < st) sred[tid] += sred[tid + st]; __syncthreads(); }
  float mu = sred[0] / E; __syncthreads();
  float v2 = 0.f;
  for (int i = tid; i < E; i += 256) { float d = xr[i] - mu; v2 += d * d; }
  sred[tid] = v2; __syncthreads();
  for (int st = 128; st > 0; st >>= 1) { if (tid < st) sred[tid] += sred[tid + st]; __syncthreads(); }
  float rstd = rsqrtf(sred[0] / E + 1e-5f);
  float* orow = out + (size_t)row * E;
  for (int i = tid; i < E; i += 256) orow[i] = (xr[i] - mu) * rstd * w[i] + b[i];
}

__global__ void scores_k(const float* __restrict__ q, const float* __restrict__ k,
                         float* __restrict__ sc, int Ntok, int E, int hd, float scale) {
  int bh = blockIdx.x;                 // 0 .. BATCH*NHEAD-1
  int b = bh / NHEAD, h = bh % NHEAD;
  int tid = blockIdx.y * 256 + threadIdx.x;
  if (tid >= Ntok * Ntok) return;
  int qi = tid / Ntok, ki = tid % Ntok;
  const float* qp = q + ((size_t)b * Ntok + qi) * E + h * hd;
  const float* kp = k + ((size_t)b * Ntok + ki) * E + h * hd;
  float d = 0.f;
  for (int e = 0; e < hd; ++e) d += qp[e] * kp[e];
  sc[(size_t)bh * Ntok * Ntok + (size_t)qi * Ntok + ki] = d * scale;
}

// softmax over the *query* axis (reference uses axis=-2)
__global__ void softmaxq_k(float* __restrict__ sc, int Ntok, int total) {
  int tid = blockIdx.x * 256 + threadIdx.x;
  if (tid >= total) return;
  int bh = tid / Ntok, kc = tid % Ntok;
  float* col = sc + (size_t)bh * Ntok * Ntok + kc;
  float mx = -1e30f;
  for (int qv = 0; qv < Ntok; ++qv) mx = fmaxf(mx, col[(size_t)qv * Ntok]);
  float sum = 0.f;
  for (int qv = 0; qv < Ntok; ++qv) { float e = expf(col[(size_t)qv * Ntok] - mx); col[(size_t)qv * Ntok] = e; sum += e; }
  float inv = 1.f / sum;
  for (int qv = 0; qv < Ntok; ++qv) col[(size_t)qv * Ntok] *= inv;
}

__global__ void attnout_k(const float* __restrict__ sc, const float* __restrict__ v,
                          float* __restrict__ x, int Ntok, int E, int hd, int total) {
  int idx = blockIdx.x * 256 + threadIdx.x;
  if (idx >= total) return;
  int e = idx % E;
  int n = (idx / E) % Ntok;
  int b = idx / (E * Ntok);
  int h = e / hd, ei = e % hd;
  const float* w = sc + ((size_t)(b * NHEAD + h) * Ntok + n) * Ntok;
  const float* vp = v + (size_t)b * Ntok * E + h * hd + ei;
  float acc = 0.f;
  for (int kk = 0; kk < Ntok; ++kk) acc += w[kk] * vp[(size_t)kk * E];
  x[idx] += acc;    // residual add
}

__global__ void mask_k(const int* __restrict__ restore, float* __restrict__ maskout) {
  int idx = blockIdx.x * 256 + threadIdx.x;
  if (idx >= BATCH * TTOK) return;
  maskout[idx] = (restore[idx] >= KEEPN) ? 1.f : 0.f;
}

__global__ void zero2_k(float* __restrict__ a) { a[0] = 0.f; a[1] = 0.f; }

__global__ void loss_k(const float* __restrict__ pred, const float* __restrict__ tgt,
                       const float* __restrict__ mask, float* __restrict__ accum) {
  __shared__ float sred[256];
  int bt = blockIdx.x;                  // 0 .. BATCH*TTOK-1
  int b = bt / TTOK, t = bt % TTOK;
  int tid = threadIdx.x;
  const float* pr = pred + ((size_t)b * DEC_N + t + 1) * PDIM;
  const float* tg = tgt + (size_t)bt * PDIM;
  float s = 0.f;
  for (int i = tid; i < PDIM; i += 256) { float d = pr[i] - tg[i]; s += d * d; }
  sred[tid] = s; __syncthreads();
  for (int st = 128; st > 0; st >>= 1) { if (tid < st) sred[tid] += sred[tid + st]; __syncthreads(); }
  if (tid == 0) {
    float m = mask[bt];
    atomicAdd(&accum[0], (sred[0] / (float)PDIM) * m);
    atomicAdd(&accum[1], m);
  }
}

__global__ void predcopy_k(const float* __restrict__ predf, float* __restrict__ out) {
  int idx = blockIdx.x * 256 + threadIdx.x;
  if (idx >= BATCH * TTOK * PDIM) return;
  int c = idx % PDIM;
  int t = (idx / PDIM) % TTOK;
  int b = idx / (PDIM * TTOK);
  out[idx] = predf[((size_t)b * DEC_N + t + 1) * PDIM + c];
}

__global__ void finalize_k(const float* __restrict__ accum, float* __restrict__ out) {
  out[0] = accum[0] / accum[1];
}

// ---------------------------------------------------------------------------
// Host-side orchestration
// ---------------------------------------------------------------------------
static inline unsigned cdiv(size_t a, unsigned b) { return (unsigned)((a + b - 1) / b); }

static void gemmL(hipStream_t s, const float* A, const float* B, const float* bias,
                  float* C, int M, int N, int K, int transB, int addC) {
  dim3 g(cdiv(N, 64), cdiv(M, 64));
  gemm_bf16_wmma<<<g, 128, 0, s>>>(A, B, bias, C, M, N, K, transB, addC);
}

static void run_layer(hipStream_t s, float* x, float* h, float* q, float* k, float* v,
                      float* scores, float* mid, int Ntok, int E, int hd, int FF,
                      const float* lnw, const float* lnb,
                      const float* qw, const float* qb, const float* kw, const float* kb,
                      const float* vw, const float* vb,
                      const float* f1w, const float* f1b,
                      const float* f2w, const float* f2b) {
  const int M = BATCH * Ntok;
  layernorm_k<<<M, 256, 0, s>>>(x, lnw, lnb, h, E);
  gemmL(s, h, qw, qb, q, M, E, E, 0, 0);
  gemmL(s, h, kw, kb, k, M, E, E, 0, 0);
  gemmL(s, h, vw, vb, v, M, E, E, 0, 0);
  const int npairs = Ntok * Ntok;
  scores_k<<<dim3(BATCH * NHEAD, cdiv(npairs, 256)), 256, 0, s>>>(q, k, scores, Ntok, E, hd, 0.25f);
  softmaxq_k<<<cdiv(BATCH * NHEAD * Ntok, 256), 256, 0, s>>>(scores, Ntok, BATCH * NHEAD * Ntok);
  attnout_k<<<cdiv((size_t)M * E, 256), 256, 0, s>>>(scores, v, x, Ntok, E, hd, M * E);
  layernorm_k<<<M, 256, 0, s>>>(x, lnw, lnb, h, E);
  gemmL(s, h, f1w, f1b, mid, M, FF, E, 0, 0);
  gemmL(s, mid, f2w, f2b, x, M, E, FF, 0, 1);   // x += FFN(h)
}

extern "C" void kernel_launch(void* const* d_in, const int* in_sizes, int n_in,
                              void* d_out, int out_size, void* d_ws, size_t ws_size,
                              hipStream_t stream) {
  const float* img        = (const float*)d_in[0];
  const float* conv_w     = (const float*)d_in[1];
  const float* conv_b     = (const float*)d_in[2];
  const float* cls_token  = (const float*)d_in[3];
  const float* pos_embed  = (const float*)d_in[4];
  const float* enc_ln_w   = (const float*)d_in[5];
  const float* enc_ln_b   = (const float*)d_in[6];
  const float* enc_qw     = (const float*)d_in[7];
  const float* enc_qb     = (const float*)d_in[8];
  const float* enc_kw     = (const float*)d_in[9];
  const float* enc_kb     = (const float*)d_in[10];
  const float* enc_vw     = (const float*)d_in[11];
  const float* enc_vb     = (const float*)d_in[12];
  const float* enc_f1w    = (const float*)d_in[13];
  const float* enc_f1b    = (const float*)d_in[14];
  const float* enc_f2w    = (const float*)d_in[15];
  const float* enc_f2b    = (const float*)d_in[16];
  const float* dec_emb_w  = (const float*)d_in[17];
  const float* dec_emb_b  = (const float*)d_in[18];
  const float* mask_token = (const float*)d_in[19];
  const float* dec_ln_w   = (const float*)d_in[20];
  const float* dec_ln_b   = (const float*)d_in[21];
  const float* dec_qw     = (const float*)d_in[22];
  const float* dec_qb     = (const float*)d_in[23];
  const float* dec_kw     = (const float*)d_in[24];
  const float* dec_kb     = (const float*)d_in[25];
  const float* dec_vw     = (const float*)d_in[26];
  const float* dec_vb     = (const float*)d_in[27];
  const float* dec_f1w    = (const float*)d_in[28];
  const float* dec_f1b    = (const float*)d_in[29];
  const float* dec_f2w    = (const float*)d_in[30];
  const float* dec_f2b    = (const float*)d_in[31];
  const float* pred_w     = (const float*)d_in[32];
  const float* pred_b     = (const float*)d_in[33];
  const float* proj_w     = (const float*)d_in[34];
  const float* proj_b     = (const float*)d_in[35];
  const int*   ids_shuf   = (const int*)d_in[36];

  // ---- workspace carve-out (floats) ----
  float* ws = (float*)d_ws;
  size_t off = 0;
  auto carve = [&](size_t n) { float* p = ws + off; off += n; return p; };
  float* patches = carve((size_t)BATCH * TTOK * PDIM);
  float* embed   = carve((size_t)BATCH * TTOK * EMB);
  float* x       = carve((size_t)BATCH * DEC_N * EMB);   // shared enc/dec stream
  float* h       = carve((size_t)BATCH * DEC_N * EMB);
  float* q       = carve((size_t)BATCH * DEC_N * EMB);
  float* k       = carve((size_t)BATCH * DEC_N * EMB);
  float* v       = carve((size_t)BATCH * DEC_N * EMB);
  float* scores  = carve((size_t)BATCH * NHEAD * DEC_N * DEC_N);
  float* mid     = carve((size_t)BATCH * DEC_N * 2048 > (size_t)BATCH * ENC_N * 4096
                         ? (size_t)BATCH * DEC_N * 2048 : (size_t)BATCH * ENC_N * 4096);
  float* ydec    = carve((size_t)BATCH * ENC_N * DECD);
  float* predf   = carve((size_t)BATCH * DEC_N * PDIM);
  float* target  = carve((size_t)BATCH * TTOK * PDIM);
  int*   idsr    = (int*)(ws + off); off += BATCH * TTOK;
  float* accum   = carve(2);
  (void)ws_size; (void)in_sizes; (void)n_in; (void)out_size;

  float* out_f    = (float*)d_out;
  float* out_pred = out_f + 1;
  float* out_mask = out_f + 1 + (size_t)BATCH * TTOK * PDIM;

  // 1) patchify + patch embed (x = patches @ conv_w^T + conv_b)
  patchify_k<<<cdiv((size_t)BATCH * TTOK * PDIM, 256), 256, 0, stream>>>(img, patches);
  gemmL(stream, patches, conv_w, conv_b, embed, BATCH * TTOK, EMB, PDIM, /*transB=*/1, 0);

  // 2) masking bookkeeping + encoder input assembly
  idsrestore_k<<<cdiv(BATCH * TTOK, 256), 256, 0, stream>>>(ids_shuf, idsr);
  enc_assemble_k<<<cdiv((size_t)BATCH * ENC_N * EMB, 256), 256, 0, stream>>>(
      embed, cls_token, pos_embed, ids_shuf, x);

  // 3) encoder: 24 layers, E=1024, hd=64, FF=4096
  for (int l = 0; l < ENC_L; ++l) {
    run_layer(stream, x, h, q, k, v, scores, mid, ENC_N, EMB, EMB / NHEAD, 4 * EMB,
              enc_ln_w + (size_t)l * EMB, enc_ln_b + (size_t)l * EMB,
              enc_qw + (size_t)l * EMB * EMB, enc_qb + (size_t)l * EMB,
              enc_kw + (size_t)l * EMB * EMB, enc_kb + (size_t)l * EMB,
              enc_vw + (size_t)l * EMB * EMB, enc_vb + (size_t)l * EMB,
              enc_f1w + (size_t)l * EMB * 4 * EMB, enc_f1b + (size_t)l * 4 * EMB,
              enc_f2w + (size_t)l * 4 * EMB * EMB, enc_f2b + (size_t)l * EMB);
  }

  // 4) decoder embed + unshuffle with mask tokens
  gemmL(stream, x, dec_emb_w, dec_emb_b, ydec, BATCH * ENC_N, DECD, EMB, 0, 0);
  dec_assemble_k<<<cdiv((size_t)BATCH * DEC_N * DECD, 256), 256, 0, stream>>>(
      ydec, mask_token, idsr, x);

  // 5) decoder: 8 layers, E=512, hd=32, FF=2048
  for (int l = 0; l < DEC_L; ++l) {
    run_layer(stream, x, h, q, k, v, scores, mid, DEC_N, DECD, DECD / NHEAD, 4 * DECD,
              dec_ln_w + (size_t)l * DECD, dec_ln_b + (size_t)l * DECD,
              dec_qw + (size_t)l * DECD * DECD, dec_qb + (size_t)l * DECD,
              dec_kw + (size_t)l * DECD * DECD, dec_kb + (size_t)l * DECD,
              dec_vw + (size_t)l * DECD * DECD, dec_vb + (size_t)l * DECD,
              dec_f1w + (size_t)l * DECD * 4 * DECD, dec_f1b + (size_t)l * 4 * DECD,
              dec_f2w + (size_t)l * 4 * DECD * DECD, dec_f2b + (size_t)l * DECD);
  }

  // 6) heads: pred = x @ pred_w + pred_b ; target = patches @ proj_w + proj_b
  gemmL(stream, x, pred_w, pred_b, predf, BATCH * DEC_N, PDIM, DECD, 0, 0);
  gemmL(stream, patches, proj_w, proj_b, target, BATCH * TTOK, PDIM, PDIM, 0, 0);

  // 7) mask, loss, outputs
  mask_k<<<cdiv(BATCH * TTOK, 256), 256, 0, stream>>>(idsr, out_mask);
  zero2_k<<<1, 1, 0, stream>>>(accum);
  loss_k<<<BATCH * TTOK, 256, 0, stream>>>(predf, target, out_mask, accum);
  predcopy_k<<<cdiv((size_t)BATCH * TTOK * PDIM, 256), 256, 0, stream>>>(predf, out_pred);
  finalize_k<<<1, 1, 0, stream>>>(accum, out_f);
}
